// GTACM_81458349736104
// MI455X (gfx1250) — matrix operand order repaired and checked by reading, not compiled, not run on hardware
//
#include <hip/hip_runtime.h>
#include <hip/hip_bf16.h>
#include <math.h>

// ---------------------------------------------------------------------------
// Problem constants (from reference): B=32, I=512, T=2048, H=1024, O=512, NF=16
// ---------------------------------------------------------------------------
#define CB 32
#define CI 512
#define CT 2048
#define CH 1024
#define CO 512
#define CNF 16
#define CR (CB * CT)   // 65536 rows

typedef __attribute__((ext_vector_type(16))) _Float16 v16h;
typedef __attribute__((ext_vector_type(8)))  _Float16 v8h;
typedef __attribute__((ext_vector_type(8)))  float    v8f;

// ---------------------------------------------------------------------------
// CDNA5 async global->LDS helpers (ASYNCcnt-tracked DMA, ISA ch.10 / ch.08)
// ---------------------------------------------------------------------------
__device__ __forceinline__ void async_wait_le4() {
#if __has_builtin(__builtin_amdgcn_s_wait_asynccnt)
  __builtin_amdgcn_s_wait_asynccnt(4);
#else
  asm volatile("s_wait_asynccnt 4" ::: "memory");
#endif
}
__device__ __forceinline__ void async_wait_zero() {
#if __has_builtin(__builtin_amdgcn_s_wait_asynccnt)
  __builtin_amdgcn_s_wait_asynccnt(0);
#else
  asm volatile("s_wait_asynccnt 0" ::: "memory");
#endif
}
// GVS mode: 64-bit uniform SGPR base + per-lane 32-bit byte offset.
// %0 = LDS byte address (VDST), %1 = global byte offset (VADDR), %2 = SGPR base.
__device__ __forceinline__ void async_b128(unsigned lds_addr, unsigned goff,
                                           const void* sbase) {
  asm volatile("global_load_async_to_lds_b128 %0, %1, %2"
               :: "v"(lds_addr), "v"(goff), "s"(sbase) : "memory");
}

// Branch-free erf (Abramowitz-Stegun 7.1.26, |err| < 1.5e-7)
__device__ __forceinline__ float erf_bf(float x) {
  float ax = fabsf(x);
  float t  = __builtin_amdgcn_rcpf(1.0f + 0.3275911f * ax);
  float y  = t * (0.254829592f +
             t * (-0.284496736f +
             t * (1.421413741f +
             t * (-1.453152027f +
             t * 1.061405429f))));
  float r  = 1.0f - y * __expf(-ax * ax);
  return copysignf(r, x);
}

// ---------------------------------------------------------------------------
// Weight transpose + f32->f16: Wt[n][k] = (f16) W[k][n]
// ---------------------------------------------------------------------------
__global__ __launch_bounds__(256)
void transpose_w_kernel(const float* __restrict__ W, _Float16* __restrict__ Wt,
                        int K, int N) {
  int idx = blockIdx.x * 256 + threadIdx.x;
  if (idx >= K * N) return;
  int k = idx / N, n = idx % N;
  Wt[(size_t)n * K + k] = (_Float16)W[idx];
}

// ---------------------------------------------------------------------------
// temporal_mix[t][i] = b_fp[i] + sum_f sin(tn*freqs[f])*W_fp[f][i]
//                               + cos(tn*freqs[f])*W_fp[NF+f][i]
// ---------------------------------------------------------------------------
__global__ __launch_bounds__(256)
void tmix_kernel(const float* __restrict__ freqs, const float* __restrict__ W_fp,
                 const float* __restrict__ b_fp, float* __restrict__ tmix) {
  int idx = blockIdx.x * 256 + threadIdx.x;     // T*I elements
  int t = idx >> 9;                             // /512
  int i = idx & 511;
  float tn = (float)t * (1.0f / (float)(CT - 1));
  float acc = b_fp[i];
#pragma unroll 4
  for (int f = 0; f < CNF; ++f) {
    float s, c;
    __sincosf(tn * freqs[f], &s, &c);
    acc += s * W_fp[f * CI + i] + c * W_fp[(CNF + f) * CI + i];
  }
  tmix[idx] = acc;
}

// ---------------------------------------------------------------------------
// Stage A: transpose x[B,I,T] -> rows, LayerNorm over I, add temporal mix,
// write f16 A[R][I]. One workgroup handles (b, 16 consecutive t's).
// ---------------------------------------------------------------------------
__global__ __launch_bounds__(256)
void stage_a_kernel(const float* __restrict__ x, const float* __restrict__ gamma,
                    const float* __restrict__ beta, const float* __restrict__ tmix,
                    _Float16* __restrict__ A) {
  __shared__ float xs[CI * 17];          // [i][t], padded stride 17
  __shared__ float psum[16 * 16];
  __shared__ float psq[16 * 16];
  __shared__ float smu[16];
  __shared__ float srs[16];

  int b  = blockIdx.y;
  int t0 = blockIdx.x * 16;
  int tid = threadIdx.x;

  const float* xb = x + (size_t)b * CI * CT;
#pragma unroll
  for (int l = 0; l < 32; ++l) {
    int flat = l * 256 + tid;            // 0..8191
    int i = flat >> 4;
    int t = flat & 15;
    xs[i * 17 + t] = xb[(size_t)i * CT + t0 + t];
  }
  __syncthreads();

  {
    int t = tid & 15, part = tid >> 4;
    float s = 0.f, q = 0.f;
#pragma unroll 8
    for (int ii = 0; ii < 32; ++ii) {
      float v = xs[(part * 32 + ii) * 17 + t];
      s += v; q += v * v;
    }
    psum[t * 16 + part] = s;
    psq [t * 16 + part] = q;
  }
  __syncthreads();
  if (tid < 16) {
    float s = 0.f, q = 0.f;
#pragma unroll
    for (int p = 0; p < 16; ++p) { s += psum[tid * 16 + p]; q += psq[tid * 16 + p]; }
    float mu  = s * (1.0f / (float)CI);
    float var = q * (1.0f / (float)CI) - mu * mu;
    smu[tid] = mu;
    srs[tid] = rsqrtf(var + 1e-5f);
  }
  __syncthreads();

#pragma unroll
  for (int l = 0; l < 32; ++l) {
    int flat = l * 256 + tid;
    int t = flat >> 9;                   // 0..15
    int i = flat & 511;
    float v = (xs[i * 17 + t] - smu[t]) * srs[t] * gamma[i] + beta[i]
            + tmix[(size_t)(t0 + t) * CI + i];
    A[((size_t)(b * CT + t0 + t)) * CI + i] = (_Float16)v;
  }
}

// ---------------------------------------------------------------------------
// WMMA GEMM: C[M][N] = act( A[M][K] @ Bt[N][K]^T + bias[N] )
// Block tile 128x128, 8 waves of 32x64, K-step 32.
// Double-buffered: async global->LDS DMA of tile k+1 overlaps WMMA on tile k.
// ACT: 0 = exact GELU, 1 = scale by (1 + *sPtr), 2 = sigmoid
// ---------------------------------------------------------------------------
#define LDAH 40            // padded LDS row stride in halves (80 bytes)
#define BUFH (128 * LDAH)  // halves per buffer

template <int ACT>
__global__ __launch_bounds__(256)
void gemm_kernel(const _Float16* __restrict__ A, const _Float16* __restrict__ Bt,
                 const float* __restrict__ bias, _Float16* __restrict__ C,
                 int M, int N, int K, const float* __restrict__ sPtr) {
  __shared__ _Float16 shA[2][BUFH];
  __shared__ _Float16 shB[2][BUFH];

  const int bm = blockIdx.y * 128;
  const int bn = blockIdx.x * 128;
  const int tid  = threadIdx.x;
  const int wave = tid >> 5;
  const int lane = tid & 31;
  const int wm = (wave & 3) * 32;        // 4 waves along M
  const int wn = (wave >> 2) * 64;       // 2 waves along N

  v8f acc[2][4] = {};

  // per-thread tile slot: row ldr (0..127), starting half ldc (0 or 16)
  const int ldr = tid >> 1;
  const int ldc = (tid & 1) * 16;
  const unsigned gv   = (unsigned)((ldr * K + ldc) * 2);   // global byte offset in tile
  // Flat LDS pointers: low 32 bits == LDS byte offset (ISA 10.2 aperture mapping)
  const unsigned laB  = (unsigned)(uintptr_t)(&shA[0][0]);
  const unsigned lbB  = (unsigned)(uintptr_t)(&shB[0][0]);
  const unsigned slot = (unsigned)((ldr * LDAH + ldc) * 2);

  const int nk = K >> 5;

  auto issue_tile = [&](int kt, int buf) {
    const _Float16* sa = A  + (size_t)bm * K + kt * 32;
    const _Float16* sb = Bt + (size_t)bn * K + kt * 32;
    unsigned la = laB + (unsigned)(buf * BUFH * 2) + slot;
    unsigned lb = lbB + (unsigned)(buf * BUFH * 2) + slot;
    async_b128(la,      gv,      sa);
    async_b128(la + 16, gv + 16, sa);
    async_b128(lb,      gv,      sb);
    async_b128(lb + 16, gv + 16, sb);
  };

  issue_tile(0, 0);

  const int mrow = lane & 15;
  const int hi   = lane >> 4;            // 0 or 1

  for (int kt = 0; kt < nk; ++kt) {
    const int cur = kt & 1;
    if (kt + 1 < nk) {
      issue_tile(kt + 1, cur ^ 1);       // DMA next tile into other buffer
      async_wait_le4();                  // in-order: <=4 outstanding => tile kt done
    } else {
      async_wait_zero();
    }
    __syncthreads();                     // all waves' DMA for tile kt visible

    const _Float16* Ac = &shA[cur][0];
    const _Float16* Bc = &shB[cur][0];

    v16h afr[2];
    v16h bfr[4];
#pragma unroll
    for (int i = 0; i < 2; ++i) {
      // A 16x32 f16: lane<16 holds K 0..7 & 16..23; lane>=16 holds K 8..15 & 24..31
      const _Float16* p = &Ac[(wm + i * 16 + mrow) * LDAH];
      v8h lo = *(const v8h*)(p + hi * 8);
      v8h hh = *(const v8h*)(p + 16 + hi * 8);
      afr[i] = __builtin_shufflevector(lo, hh, 0,1,2,3,4,5,6,7,8,9,10,11,12,13,14,15);
    }
#pragma unroll
    for (int j = 0; j < 4; ++j) {
      // B 32x16 f16: lane%16 = column N, lane<16 -> K 0..7 & 16..23, lane>=16 -> K 8..15 & 24..31
      const _Float16* p = &Bc[(wn + j * 16 + mrow) * LDAH];
      v8h lo = *(const v8h*)(p + hi * 8);
      v8h hh = *(const v8h*)(p + 16 + hi * 8);
      bfr[j] = __builtin_shufflevector(lo, hh, 0,1,2,3,4,5,6,7,8,9,10,11,12,13,14,15);
    }

#pragma unroll
    for (int i = 0; i < 2; ++i)
#pragma unroll
      for (int j = 0; j < 4; ++j)
        acc[i][j] = __builtin_amdgcn_wmma_f32_16x16x32_f16(
            false, afr[i], false, bfr[j], (short)0, acc[i][j], false, false);

    __syncthreads();                     // all waves done reading buffer cur
  }

  // ---- epilogue: C layout  lane<16: N=lane, M=r ; lane>=16: N=lane-16, M=8+r
  float sc = 1.0f;
  if constexpr (ACT == 1) sc = 1.0f + sPtr[0];
  const int ncol  = lane & 15;
  const int mbase = (lane >> 4) * 8;
#pragma unroll
  for (int i = 0; i < 2; ++i) {
#pragma unroll
    for (int j = 0; j < 4; ++j) {
      int col = bn + wn + j * 16 + ncol;
      float bv = bias[col];
#pragma unroll
      for (int r = 0; r < 8; ++r) {
        int row = bm + wm + i * 16 + mbase + r;
        float v = acc[i][j][r] + bv;
        if constexpr (ACT == 0) {
          v = 0.5f * v * (1.0f + erf_bf(v * 0.70710678118f));   // exact GELU
        } else if constexpr (ACT == 1) {
          v = v * sc;
        } else {
          v = __builtin_amdgcn_rcpf(1.0f + __expf(-v));          // sigmoid
        }
        C[(size_t)row * N + col] = (_Float16)v;
      }
    }
  }
}

// ---------------------------------------------------------------------------
// Softmax over H per row + attention pooling, deterministic partials.
// WG handles 32 rows (one b, one t-chunk). partial[b][chunk][H].
// ---------------------------------------------------------------------------
__global__ __launch_bounds__(256)
void softmax_pool_kernel(const _Float16* __restrict__ gate,
                         const _Float16* __restrict__ h,
                         float* __restrict__ partial) {
  __shared__ float lds[8 * CH];          // [group][col], 32 KB

  const int chunk = blockIdx.x;          // 0 .. R/32-1
  const int row0  = chunk * 32;
  const int b     = row0 / CT;
  const int lchunk = chunk - b * (CT / 32);
  const int lane = threadIdx.x & 31;
  const int grp  = threadIdx.x >> 5;     // 0..7

  float pacc[32];
#pragma unroll
  for (int j = 0; j < 32; ++j) pacc[j] = 0.f;

  for (int rr = 0; rr < 4; ++rr) {
    int row = row0 + rr * 8 + grp;
    const _Float16* g  = gate + (size_t)row * CH;
    const _Float16* hr = h    + (size_t)row * CH;
    float gv[32];
    float mx = -1e30f;
#pragma unroll
    for (int j = 0; j < 32; ++j) {
      float v = (float)g[lane + 32 * j];
      gv[j] = v;
      mx = fmaxf(mx, v);
    }
#pragma unroll
    for (int off = 16; off > 0; off >>= 1) mx = fmaxf(mx, __shfl_xor(mx, off, 32));
    float s = 0.f;
#pragma unroll
    for (int j = 0; j < 32; ++j) { float e = __expf(gv[j] - mx); gv[j] = e; s += e; }
#pragma unroll
    for (int off = 16; off > 0; off >>= 1) s += __shfl_xor(s, off, 32);
    float inv = __builtin_amdgcn_rcpf(s);
#pragma unroll
    for (int j = 0; j < 32; ++j)
      pacc[j] += gv[j] * inv * (float)hr[lane + 32 * j];
  }

#pragma unroll
  for (int j = 0; j < 32; ++j) lds[grp * CH + lane + 32 * j] = pacc[j];
  __syncthreads();
  for (int c = threadIdx.x; c < CH; c += 256) {
    float s = 0.f;
#pragma unroll
    for (int g2 = 0; g2 < 8; ++g2) s += lds[g2 * CH + c];
    partial[((size_t)b * (CT / 32) + lchunk) * CH + c] = s;
  }
}

__global__ __launch_bounds__(256)
void pool_reduce_kernel(const float* __restrict__ partial, float* __restrict__ pooled) {
  int idx = blockIdx.x * 256 + threadIdx.x;   // B*H
  int b = idx >> 10;
  int c = idx & 1023;
  float s = 0.f;
  for (int k = 0; k < CT / 32; ++k)
    s += partial[((size_t)b * (CT / 32) + k) * CH + c];
  pooled[idx] = s;
}

// ---------------------------------------------------------------------------
// Final tiny GEMM: out[b][o] = b_out[o] + sum_h pooled[b][h] * W_out[h][o]
// ---------------------------------------------------------------------------
__global__ __launch_bounds__(256)
void out_kernel(const float* __restrict__ pooled, const float* __restrict__ W_out,
                const float* __restrict__ b_out, float* __restrict__ out) {
  int idx = blockIdx.x * 256 + threadIdx.x;   // B*O
  int b = idx >> 9;
  int o = idx & 511;
  float acc = b_out[o];
  const float* p = pooled + (size_t)b * CH;
#pragma unroll 8
  for (int hh = 0; hh < CH; ++hh) acc += p[hh] * W_out[(size_t)hh * CO + o];
  out[idx] = acc;
}

// ---------------------------------------------------------------------------
// Host-side launch
// ---------------------------------------------------------------------------
extern "C" void kernel_launch(void* const* d_in, const int* in_sizes, int n_in,
                              void* d_out, int out_size, void* d_ws, size_t ws_size,
                              hipStream_t stream) {
  (void)in_sizes; (void)n_in; (void)out_size; (void)ws_size;

  const float* x        = (const float*)d_in[0];
  const float* ln_gamma = (const float*)d_in[1];
  const float* ln_beta  = (const float*)d_in[2];
  const float* freqs    = (const float*)d_in[3];
  const float* W_fp     = (const float*)d_in[4];
  const float* b_fp     = (const float*)d_in[5];
  const float* W_c1     = (const float*)d_in[6];
  const float* b_c1     = (const float*)d_in[7];
  const float* W_c2     = (const float*)d_in[8];
  const float* b_c2     = (const float*)d_in[9];
  const float* res_sc   = (const float*)d_in[10];
  const float* W_g1     = (const float*)d_in[11];
  const float* b_g1     = (const float*)d_in[12];
  const float* W_g2     = (const float*)d_in[13];
  const float* b_g2     = (const float*)d_in[14];
  const float* W_out    = (const float*)d_in[15];
  const float* b_out    = (const float*)d_in[16];
  float* out = (float*)d_out;

  // ---- workspace layout ----
  char* ws = (char*)d_ws;
  size_t off = 0;
  auto carve = [&](size_t bytes) -> void* {
    void* p = ws + off;
    off += (bytes + 255) & ~(size_t)255;
    return p;
  };
  _Float16* Wc1t   = (_Float16*)carve((size_t)CH * CI * 2);   // [H][I]
  _Float16* Wc2t   = (_Float16*)carve((size_t)CH * CH * 2);   // [H][H]
  _Float16* Wg1t   = (_Float16*)carve((size_t)CH * CH * 2);
  _Float16* Wg2t   = (_Float16*)carve((size_t)CH * CH * 2);
  float*    tmix   = (float*)   carve((size_t)CT * CI * 4);   // [T][I]
  float*    pooled = (float*)   carve((size_t)CB * CH * 4);
  float*    partial= (float*)   carve((size_t)CB * (CT / 32) * CH * 4);
  _Float16* bufA   = (_Float16*)carve((size_t)CR * CH * 2);   // A [R][I] then G1 [R][H]
  _Float16* bufB   = (_Float16*)carve((size_t)CR * CH * 2);   // H1 then gate
  _Float16* bufC   = (_Float16*)carve((size_t)CR * CH * 2);   // h (scaled)

  // ---- 1. weight prep (f32 -> f16 transposed) ----
  transpose_w_kernel<<<(CI * CH + 255) / 256, 256, 0, stream>>>(W_c1, Wc1t, CI, CH);
  transpose_w_kernel<<<(CH * CH + 255) / 256, 256, 0, stream>>>(W_c2, Wc2t, CH, CH);
  transpose_w_kernel<<<(CH * CH + 255) / 256, 256, 0, stream>>>(W_g1, Wg1t, CH, CH);
  transpose_w_kernel<<<(CH * CH + 255) / 256, 256, 0, stream>>>(W_g2, Wg2t, CH, CH);

  // ---- 2. temporal mix table ----
  tmix_kernel<<<(CT * CI) / 256, 256, 0, stream>>>(freqs, W_fp, b_fp, tmix);

  // ---- 3. layernorm + transpose + mix -> A f16 ----
  stage_a_kernel<<<dim3(CT / 16, CB), 256, 0, stream>>>(x, ln_gamma, ln_beta, tmix, bufA);

  // ---- 4-7. WMMA GEMM chain ----
  dim3 ggrid(CH / 128, CR / 128);
  gemm_kernel<0><<<ggrid, 256, 0, stream>>>(bufA, Wc1t, b_c1, bufB, CR, CH, CI, nullptr);
  gemm_kernel<1><<<ggrid, 256, 0, stream>>>(bufB, Wc2t, b_c2, bufC, CR, CH, CH, res_sc);
  gemm_kernel<0><<<ggrid, 256, 0, stream>>>(bufC, Wg1t, b_g1, bufA, CR, CH, CH, nullptr);
  gemm_kernel<2><<<ggrid, 256, 0, stream>>>(bufA, Wg2t, b_g2, bufB, CR, CH, CH, nullptr);

  // ---- 8. softmax over H + attention pooling (deterministic) ----
  softmax_pool_kernel<<<CR / 32, 256, 0, stream>>>(bufB, bufC, partial);
  pool_reduce_kernel<<<(CB * CH) / 256, 256, 0, stream>>>(partial, pooled);

  // ---- 9. output projection ----
  out_kernel<<<(CB * CO) / 256, 256, 0, stream>>>(pooled, W_out, b_out, out);
}
